// EncoderBlock_8890582302802
// MI455X (gfx1250) — compile-verified, matrix operand order checked by reading, not run on hardware
//
#include <hip/hip_runtime.h>
#include <hip/hip_bf16.h>

#define NB  4
#define LL  2048
#define DD  1024
#define DI  2048
#define NS  16
#define DTR 64
#define MM  (NB * LL)   // 8192 rows
#define XDS 128         // padded x_dbl stride

typedef __attribute__((ext_vector_type(16))) __bf16 v16bf;
typedef __attribute__((ext_vector_type(8)))  __bf16 v8bf;
typedef __attribute__((ext_vector_type(8)))  float  v8f;
typedef __attribute__((ext_vector_type(4)))  unsigned int u32x4;
typedef __attribute__((ext_vector_type(8)))  int i32x8;
typedef __attribute__((ext_vector_type(4)))  int i32x4;

__device__ __forceinline__ unsigned short f2bf(float f) {
    unsigned int u = __float_as_uint(f);
    u += 0x7FFFu + ((u >> 16) & 1u);          // round-to-nearest-even
    return (unsigned short)(u >> 16);
}
__device__ __forceinline__ float silu_f(float x) { return x / (1.f + __expf(-x)); }

// ---------------- generic f32 -> bf16 cast ----------------
__global__ void k_cast_bf16(const float* __restrict__ src, unsigned short* __restrict__ dst, int n) {
    int i = blockIdx.x * 256 + threadIdx.x;
    if (i < n) dst[i] = f2bf(src[i]);
}

// ---------------- x_proj_w cast with zero-pad 96 -> 128 rows ----------------
__global__ void k_cast_pad_wx(const float* __restrict__ src, unsigned short* __restrict__ dst) {
    int i = blockIdx.x * 256 + threadIdx.x;   // over 128*DI
    if (i >= XDS * DI) return;
    int n = i / DI, k = i % DI;
    dst[i] = (n < 96) ? f2bf(src[n * DI + k]) : (unsigned short)0;
}

// ---------------- RMSNorm + cast ----------------
__global__ void k_rmsnorm(const float* __restrict__ x, const float* __restrict__ w,
                          unsigned short* __restrict__ xb) {
    int row = blockIdx.x;                       // 0..MM-1
    const float* xr = x + (size_t)row * DD;
    __shared__ float red[256];
    float s = 0.f;
    for (int d = threadIdx.x; d < DD; d += 256) { float v = xr[d]; s += v * v; }
    red[threadIdx.x] = s; __syncthreads();
    for (int off = 128; off > 0; off >>= 1) {
        if (threadIdx.x < off) red[threadIdx.x] += red[threadIdx.x + off];
        __syncthreads();
    }
    float r = rsqrtf(red[0] / (float)DD + 1e-6f);
    for (int d = threadIdx.x; d < DD; d += 256)
        xb[(size_t)row * DD + d] = f2bf(xr[d] * r * w[d]);
}

// ---------------- TDM tile load: 128 rows x 32 bf16, row stride K ----------------
__device__ __forceinline__ void tdm_tile_load(const unsigned short* gsrc, unsigned lds_off, int K) {
#if __has_builtin(__builtin_amdgcn_tensor_load_to_lds)
    unsigned long long ga = (unsigned long long)(uintptr_t)gsrc;
    u32x4 g0;
    g0[0] = 1u;                                               // count=1, user mode
    g0[1] = lds_off;                                          // lds_addr
    g0[2] = (unsigned)ga;                                     // global_addr[31:0]
    g0[3] = (unsigned)((ga >> 32) & 0x01FFFFFFu) | (2u << 30);// global_addr[56:32] | type=2
    i32x8 g1;
    g1[0] = 0x00010000;                                       // data_size=1 (2 bytes)
    g1[1] = (int)(((unsigned)K & 0xFFFFu) << 16);             // tensor_dim0[15:0]
    g1[2] = (int)((((unsigned)K >> 16) & 0xFFFFu) | (128u << 16)); // dim0 hi | tensor_dim1 lo
    g1[3] = (int)(32u << 16);                                 // tensor_dim1 hi=0 | tile_dim0=32
    g1[4] = 128;                                              // tile_dim1=128, tile_dim2=0
    g1[5] = K;                                                // tensor_dim0_stride[31:0]
    g1[6] = 0;
    g1[7] = 0;
    i32x4 z4 = {0, 0, 0, 0};
#if defined(__clang_major__) && __clang_major__ >= 23
    i32x8 z8 = {0, 0, 0, 0, 0, 0, 0, 0};
    __builtin_amdgcn_tensor_load_to_lds(g0, g1, z4, z4, z8, 0);
#else
    __builtin_amdgcn_tensor_load_to_lds(g0, g1, z4, z4, 0);
#endif
#endif
}

// ---------------- LDS fragment load (A/B tile row-major [*,32]) ----------------
__device__ __forceinline__ v16bf frag_ld(const unsigned short* base, int row, int half) {
    const unsigned short* p = base + row * 32 + half * 8;
    v8bf lo = *(const v8bf*)p;
    v8bf hi = *(const v8bf*)(p + 16);
    v16bf r;
    #pragma unroll
    for (int i = 0; i < 8; ++i) { r[i] = lo[i]; r[i + 8] = hi[i]; }
    return r;
}

// ---------------- bf16 WMMA GEMM:  C[M,N] = A[M,K] * W[N,K]^T ----------------
// Block tile 128x128, BK=32, 256 threads = 8 waves (4M x 2N), wave tile 32x64.
// TDM double-buffered LDS staging; wave 0 drives the Tensor Data Mover.
__global__ void k_gemm_bf16(const unsigned short* __restrict__ A,
                            const unsigned short* __restrict__ W,
                            float* __restrict__ C, int M, int N, int K) {
    __shared__ __align__(16) unsigned short sA[2][128 * 32];
    __shared__ __align__(16) unsigned short sB[2][128 * 32];
    int tid  = threadIdx.x;
    int wave = tid >> 5, lane = tid & 31;
    int half = lane >> 4, l16 = lane & 15;
    int rowW = wave & 3, colW = wave >> 2;    // 4 x 2 wave grid
    int m0 = blockIdx.x * 128;
    int n0 = blockIdx.y * 128;
    v8f acc[2][4] = {};

#if __has_builtin(__builtin_amdgcn_tensor_load_to_lds)
    // prologue: stage tile k0=0 into buffer 0
    if (wave == 0) {
        tdm_tile_load(A + (size_t)m0 * K, (unsigned)(uintptr_t)&sA[0][0], K);
        tdm_tile_load(W + (size_t)n0 * K, (unsigned)(uintptr_t)&sB[0][0], K);
        __builtin_amdgcn_s_wait_tensorcnt((short)0);
    }
    __syncthreads();
    int cur = 0;
    for (int k0 = 0; k0 < K; k0 += 32) {
        if (wave == 0 && (k0 + 32) < K) {     // prefetch next tile into other buffer
            tdm_tile_load(A + (size_t)m0 * K + (k0 + 32), (unsigned)(uintptr_t)&sA[cur ^ 1][0], K);
            tdm_tile_load(W + (size_t)n0 * K + (k0 + 32), (unsigned)(uintptr_t)&sB[cur ^ 1][0], K);
        }
        const unsigned short* aB = sA[cur];
        const unsigned short* bB = sB[cur];
        v16bf af[2], bfr[4];
        #pragma unroll
        for (int mi = 0; mi < 2; ++mi) af[mi]  = frag_ld(aB, rowW * 32 + mi * 16 + l16, half);
        #pragma unroll
        for (int ni = 0; ni < 4; ++ni) bfr[ni] = frag_ld(bB, colW * 64 + ni * 16 + l16, half);
        #pragma unroll
        for (int mi = 0; mi < 2; ++mi)
            #pragma unroll
            for (int ni = 0; ni < 4; ++ni)
                acc[mi][ni] = __builtin_amdgcn_wmma_f32_16x16x32_bf16(
                    false, af[mi], false, bfr[ni], (short)0, acc[mi][ni], false, false);
        if (wave == 0) __builtin_amdgcn_s_wait_tensorcnt((short)0);
        __syncthreads();
        cur ^= 1;
    }
#else
    // fallback: synchronous cooperative staging
    for (int k0 = 0; k0 < K; k0 += 32) {
        #pragma unroll
        for (int i = 0; i < 2; ++i) {
            int c = tid + i * 256;            // A: 512 chunks of 8 bf16
            int r = c >> 2, c8 = (c & 3) * 8;
            *(uint4*)&sA[0][r * 32 + c8] = *(const uint4*)&A[(size_t)(m0 + r) * K + k0 + c8];
            *(uint4*)&sB[0][r * 32 + c8] = *(const uint4*)&W[(size_t)(n0 + r) * K + k0 + c8];
        }
        __syncthreads();
        v16bf af[2], bfr[4];
        #pragma unroll
        for (int mi = 0; mi < 2; ++mi) af[mi]  = frag_ld(sA[0], rowW * 32 + mi * 16 + l16, half);
        #pragma unroll
        for (int ni = 0; ni < 4; ++ni) bfr[ni] = frag_ld(sB[0], colW * 64 + ni * 16 + l16, half);
        #pragma unroll
        for (int mi = 0; mi < 2; ++mi)
            #pragma unroll
            for (int ni = 0; ni < 4; ++ni)
                acc[mi][ni] = __builtin_amdgcn_wmma_f32_16x16x32_bf16(
                    false, af[mi], false, bfr[ni], (short)0, acc[mi][ni], false, false);
        __syncthreads();
    }
#endif
    // C/D layout: VGPR i -> M = i + 8*half, N = l16
    #pragma unroll
    for (int mi = 0; mi < 2; ++mi) {
        int rowb = m0 + rowW * 32 + mi * 16 + half * 8;
        #pragma unroll
        for (int ni = 0; ni < 4; ++ni) {
            int col = n0 + colW * 64 + ni * 16 + l16;
            #pragma unroll
            for (int i = 0; i < 8; ++i)
                C[(size_t)(rowb + i) * N + col] = acc[mi][ni][i];
        }
    }
}

// ---------------- depthwise causal conv (K=4) + SiLU + mask ----------------
__global__ void k_conv_silu(const float* __restrict__ xz, const float* __restrict__ cw,
                            const int* __restrict__ lengths,
                            float* __restrict__ uf, unsigned short* __restrict__ ub) {
    size_t idx = (size_t)blockIdx.x * 256 + threadIdx.x;
    if (idx >= (size_t)MM * DI) return;
    int c = (int)(idx % DI);
    int l = (int)((idx / DI) % LL);
    int b = (int)(idx / ((size_t)DI * LL));
    int len = lengths[b];
    float acc = 0.f;
    #pragma unroll
    for (int k = 0; k < 4; ++k) {
        int ll = l - 3 + k;
        if (ll >= 0 && ll < len)
            acc += cw[c * 4 + k] * xz[(size_t)(b * LL + ll) * (2 * DI) + c];
    }
    float u = (l < len) ? silu_f(acc) : 0.f;
    uf[idx] = u;
    ub[idx] = f2bf(u);
}

// ---------------- extract dt_r (first 64 of x_dbl) as bf16 ----------------
__global__ void k_extract_dt(const float* __restrict__ xdbl, unsigned short* __restrict__ dt) {
    int i = blockIdx.x * 256 + threadIdx.x;
    if (i >= MM * DTR) return;
    int m = i / DTR, r = i % DTR;
    dt[i] = f2bf(xdbl[(size_t)m * XDS + r]);
}

// ---------------- softplus(delta + bias) * mask (in place) ----------------
__global__ void k_softplus(float* __restrict__ delta, const float* __restrict__ bias,
                           const int* __restrict__ lengths) {
    size_t idx = (size_t)blockIdx.x * 256 + threadIdx.x;
    if (idx >= (size_t)MM * DI) return;
    int c = (int)(idx % DI);
    int l = (int)((idx / DI) % LL);
    int b = (int)(idx / ((size_t)DI * LL));
    float x  = delta[idx] + bias[c];
    float sp = (x > 20.f) ? x : log1pf(__expf(x));
    delta[idx] = (l < lengths[b]) ? sp : 0.f;
}

// ---------------- sequential selective scan ----------------
// grid (DI/256, B); one thread per (b, channel); 8 timesteps of B/C staged per round
__global__ void k_scan(const float* __restrict__ delta, const float* __restrict__ uf,
                       const float* __restrict__ xdbl, const float* __restrict__ A_log,
                       float* __restrict__ y, float* __restrict__ h_last) {
    int b = blockIdx.y;
    int c = blockIdx.x * 256 + threadIdx.x;
    float A[NS], h[NS];
    #pragma unroll
    for (int n = 0; n < NS; ++n) { A[n] = -__expf(A_log[c * NS + n]); h[n] = 0.f; }
    __shared__ float sBC[8][32];
    int s = threadIdx.x >> 5, j = threadIdx.x & 31;
    for (int l0 = 0; l0 < LL; l0 += 8) {
        sBC[s][j] = xdbl[(size_t)(b * LL + l0 + s) * XDS + 64 + j];
        __syncthreads();
        #pragma unroll
        for (int t = 0; t < 8; ++t) {
            size_t idx = (size_t)(b * LL + l0 + t) * DI + c;
            float dlt = delta[idx];
            float du  = dlt * uf[idx];
            float yv  = 0.f;
            #pragma unroll
            for (int n = 0; n < NS; ++n) {
                h[n] = __expf(dlt * A[n]) * h[n] + du * sBC[t][n];
                yv  += h[n] * sBC[t][16 + n];
            }
            y[idx] = yv;
        }
        __syncthreads();
    }
    #pragma unroll
    for (int n = 0; n < NS; ++n)
        h_last[(size_t)(b * DI + c) * NS + n] = h[n];
}

// ---------------- y = (y + u*D_skip) * silu(z) -> bf16 ----------------
__global__ void k_gate(const float* __restrict__ y, const float* __restrict__ uf,
                       const float* __restrict__ xz, const float* __restrict__ Dsk,
                       unsigned short* __restrict__ yb) {
    size_t idx = (size_t)blockIdx.x * 256 + threadIdx.x;
    if (idx >= (size_t)MM * DI) return;
    int c = (int)(idx % DI);
    size_t row = idx / DI;
    float z  = xz[row * (2 * DI) + DI + c];
    float yy = (y[idx] + uf[idx] * Dsk[c]) * silu_f(z);
    yb[idx] = f2bf(yy);
}

// ---------------- hidden = residual + out*mask ----------------
__global__ void k_final(const float* __restrict__ hs, const float* __restrict__ outp,
                        const int* __restrict__ lengths, float* __restrict__ out) {
    size_t idx = (size_t)blockIdx.x * 256 + threadIdx.x;
    if (idx >= (size_t)MM * DD) return;
    int l = (int)((idx / DD) % LL);
    int b = (int)(idx / ((size_t)DD * LL));
    float o = (l < lengths[b]) ? outp[idx] : 0.f;
    out[idx] = hs[idx] + o;
}

extern "C" void kernel_launch(void* const* d_in, const int* in_sizes, int n_in,
                              void* d_out, int out_size, void* d_ws, size_t ws_size,
                              hipStream_t stream) {
    const float* hs        = (const float*)d_in[0];
    const int*   lengths   = (const int*)  d_in[1];
    const float* norm_w    = (const float*)d_in[2];
    const float* in_proj_w = (const float*)d_in[3];
    const float* conv_w    = (const float*)d_in[4];
    const float* x_proj_w  = (const float*)d_in[5];
    const float* dt_proj_w = (const float*)d_in[6];
    const float* dt_proj_b = (const float*)d_in[7];
    const float* A_log     = (const float*)d_in[8];
    const float* D_skip    = (const float*)d_in[9];
    const float* out_proj_w= (const float*)d_in[10];
    float* out = (float*)d_out;

    char* p = (char*)d_ws;
    auto alloc = [&](size_t bytes) -> char* {
        char* r = p; p += (bytes + 255) & ~(size_t)255; return r;
    };
    unsigned short* x_bf   = (unsigned short*)alloc((size_t)MM * DD * 2);
    unsigned short* w_in   = (unsigned short*)alloc((size_t)2 * DI * DD * 2);
    unsigned short* w_x    = (unsigned short*)alloc((size_t)XDS * DI * 2);
    unsigned short* w_dt   = (unsigned short*)alloc((size_t)DI * DTR * 2);
    unsigned short* w_out  = (unsigned short*)alloc((size_t)DD * DI * 2);
    float*          xz     = (float*)alloc((size_t)MM * 2 * DI * 4);
    float*          u_f    = (float*)alloc((size_t)MM * DI * 4);
    unsigned short* u_bf   = (unsigned short*)alloc((size_t)MM * DI * 2);
    float*          x_dbl  = (float*)alloc((size_t)MM * XDS * 4);
    unsigned short* dt_bf  = (unsigned short*)alloc((size_t)MM * DTR * 2);
    float*          delta  = (float*)alloc((size_t)MM * DI * 4);
    float*          y_f    = (float*)alloc((size_t)MM * DI * 4);
    unsigned short* y_bf   = (unsigned short*)alloc((size_t)MM * DI * 2);
    float*          out_pre= (float*)alloc((size_t)MM * DD * 4);

    // weight casts (x_proj padded 96 -> 128 rows)
    k_cast_bf16<<<(2 * DI * DD + 255) / 256, 256, 0, stream>>>(in_proj_w, w_in, 2 * DI * DD);
    k_cast_pad_wx<<<(XDS * DI + 255) / 256, 256, 0, stream>>>(x_proj_w, w_x);
    k_cast_bf16<<<(DI * DTR + 255) / 256, 256, 0, stream>>>(dt_proj_w, w_dt, DI * DTR);
    k_cast_bf16<<<(DD * DI + 255) / 256, 256, 0, stream>>>(out_proj_w, w_out, DD * DI);

    // 1. RMSNorm
    k_rmsnorm<<<MM, 256, 0, stream>>>(hs, norm_w, x_bf);
    // 2. in_proj: xz = x @ W_in^T   (M=8192, N=4096, K=1024)
    k_gemm_bf16<<<dim3(MM / 128, (2 * DI) / 128), 256, 0, stream>>>(x_bf, w_in, xz, MM, 2 * DI, DD);
    // 3. conv + silu + mask
    k_conv_silu<<<(int)(((size_t)MM * DI + 255) / 256), 256, 0, stream>>>(xz, conv_w, lengths, u_f, u_bf);
    // 4. x_proj: x_dbl = u @ W_x^T  (N=128 padded, K=2048)
    k_gemm_bf16<<<dim3(MM / 128, XDS / 128), 256, 0, stream>>>(u_bf, w_x, x_dbl, MM, XDS, DI);
    // 5. dt extraction + dt_proj (N=2048, K=64)
    k_extract_dt<<<(MM * DTR + 255) / 256, 256, 0, stream>>>(x_dbl, dt_bf);
    k_gemm_bf16<<<dim3(MM / 128, DI / 128), 256, 0, stream>>>(dt_bf, w_dt, delta, MM, DI, DTR);
    // 6. softplus(+bias)*mask
    k_softplus<<<(int)(((size_t)MM * DI + 255) / 256), 256, 0, stream>>>(delta, dt_proj_b, lengths);
    // 7. selective scan; h_last written to tail of d_out
    float* h_last = out + (size_t)MM * DD;
    k_scan<<<dim3(DI / 256, NB), 256, 0, stream>>>(delta, u_f, x_dbl, A_log, y_f, h_last);
    // 8. gating
    k_gate<<<(int)(((size_t)MM * DI + 255) / 256), 256, 0, stream>>>(y_f, u_f, xz, D_skip, y_bf);
    // 9. out_proj (N=1024, K=2048)
    k_gemm_bf16<<<dim3(MM / 128, DD / 128), 256, 0, stream>>>(y_bf, w_out, out_pre, MM, DD, DI);
    // 10. residual + mask
    k_final<<<(int)(((size_t)MM * DD + 255) / 256), 256, 0, stream>>>(hs, out_pre, lengths, out);
}
